// KANLinear_31447750542039
// MI455X (gfx1250) — matrix operand order repaired and checked by reading, not compiled, not run on hardware
//
#include <hip/hip_runtime.h>
#include <hip/hip_bf16.h>

// ---------------------------------------------------------------------------
// KAN linear layer fused as ONE bf16 WMMA GEMM:
//   out[n,o] = sum_k F[n,k] * Wc[o,k], K = 512 (silu) + 512*8 (spline) = 4608
// M=16384, N=512 -> ~77 GFLOP vs ~77MB HBM traffic: compute bound.
// Spline bases use the closed-form uniform-grid cubic B-spline (4 nonzero
// weights scattered into LDS) -- no divides, ~45 VALU per x instead of ~500.
// ---------------------------------------------------------------------------

typedef __bf16 bf16;
typedef __attribute__((ext_vector_type(16))) __bf16 v16bf;
typedef __attribute__((ext_vector_type(8)))  __bf16 v8bf;
typedef __attribute__((ext_vector_type(8)))  float  v8f;
typedef __attribute__((ext_vector_type(4)))  float  v4f;

#define IN_F   512
#define OUT_F  512
#define NCOEF  8
#define KTOT   (IN_F + IN_F * NCOEF)     // 4608
#define KSTEPS (KTOT / 32)               // 144
#define LPAD   56                        // bf16 per LDS row (112B: 16B-aligned frags)
#define CPAD   136                       // floats per LDS C row (544B)
#define ABYTES (128 * LPAD * 2)          // 14336
#define BBYTES (256 * LPAD * 2)          // 28672

// --- Kernel 0: pack combined weight [OUT_F][KTOT] to bf16 in workspace -----
__global__ __launch_bounds__(256)
void kan_pack_weights(const float* __restrict__ bw,   // (512,512)
                      const float* __restrict__ sw,   // (512,512,8)
                      const float* __restrict__ ss,   // (512,512)
                      bf16* __restrict__ Wc) {        // (512,4608)
    int idx = blockIdx.x * 256 + threadIdx.x;
    if (idx >= OUT_F * KTOT) return;
    int o = idx / KTOT;
    int k = idx - o * KTOT;
    float v;
    if (k < IN_F) {
        v = bw[o * IN_F + k];
    } else {
        int r = k - IN_F;
        int i = r >> 3, j = r & 7;
        v = sw[(o * IN_F + i) * NCOEF + j] * ss[o * IN_F + i];
    }
    Wc[(size_t)o * KTOT + k] = (bf16)v;
}

// --- Kernel 1: fused feature-map + bf16 WMMA GEMM --------------------------
// Block tile: 128 (M=rows of xt) x 256 (N=output channels), 256 threads.
// Wave tile : 32 x 128 -> acc[2][8] v8f, 16 WMMA per K-step.
__global__ __launch_bounds__(256)
void kan_main(const float* __restrict__ X,     // (8,512,2048)
              const float* __restrict__ grid,  // (512,12), rows identical
              const bf16*  __restrict__ Wc,    // (512,4608)
              float* __restrict__ Out) {       // (8,512,2048)
    __shared__ __align__(16) char smem[ABYTES + BBYTES + 16];
    bf16*  As    = (bf16*)smem;                    // [128][LPAD]
    bf16*  Bs    = (bf16*)(smem + ABYTES);         // [256][LPAD]
    float* Cs    = (float*)smem;                   // [64][CPAD] (epilogue)
    bf16*  dummy = (bf16*)(smem + ABYTES + BBYTES);// sink for dropped scatters

    const int tid  = threadIdx.x;
    const int lane = tid & 31;
    const int wave = tid >> 5;
    const int wm   = wave & 3;   // M quadrant (rows wm*32..+31)
    const int wn   = wave >> 2;  // N half    (cols wn*128..+127)

    const int m0 = blockIdx.x * 128;          // all 128 rows in one batch b
    const int n0 = blockIdx.y * 256;
    const int b  = m0 >> 11;
    const int l0 = m0 & 2047;

    // uniform spline grid: only need g[0], h, g[11]
    const float g0   = grid[0];
    const float g11  = grid[11];
    const float invh = 1.0f / (grid[1] - grid[0]);

    v8f acc[2][8];
#pragma unroll
    for (int a = 0; a < 2; ++a)
#pragma unroll
        for (int t = 0; t < 8; ++t) {
            v8f z = {0.f, 0.f, 0.f, 0.f, 0.f, 0.f, 0.f, 0.f};
            acc[a][t] = z;
        }

    // staging thread maps
    const int s_c  = tid >> 3;            // silu: channel 0..31
    const int s_r0 = (tid & 7) * 16;      // silu: 16 rows per thread
    const int q_c  = tid >> 6;            // bases: channel 0..3
    const int q_r0 = (tid & 63) * 2;      // bases: 2 rows per thread

    const int half = lane >> 4;
    const int lr   = lane & 15;

    for (int step = 0; step < KSTEPS; ++step) {
        const int kb = step * 32;
        // ---------------- A tile: 128 rows x 32 features --------------------
        if (step < 16) {
            // silu features: 32 channels, 16 consecutive l per thread
            const int ch = kb + s_c;
            const float* xp = X + (((size_t)(b * IN_F + ch)) << 11) + l0 + s_r0;
#pragma unroll
            for (int v = 0; v < 4; ++v) {
                v4f xv = *(const v4f*)(xp + v * 4);
#pragma unroll
                for (int e = 0; e < 4; ++e) {
                    float x = xv[e];
                    float s = x * __builtin_amdgcn_rcpf(1.0f + __expf(-x));
                    As[(s_r0 + v * 4 + e) * LPAD + s_c] = (bf16)s;
                }
            }
        } else {
            // cubic B-spline bases on uniform grid: 4 channels x 8 slots
            const int ch = (step - 16) * 4 + q_c;
            const float* xp = X + (((size_t)(b * IN_F + ch)) << 11) + l0 + q_r0;
            float2 xv = *(const float2*)xp;
            __builtin_prefetch(xp + (4 << 11), 0, 0);  // next step's channel
#pragma unroll
            for (int e = 0; e < 2; ++e) {
                const float x  = e ? xv.y : xv.x;
                const int  row = q_r0 + e;
                bf16* slot = &As[row * LPAD + q_c * 8];
                v8bf z8;
#pragma unroll
                for (int q = 0; q < 8; ++q) z8[q] = (bf16)0.0f;
                *(v8bf*)slot = z8;                      // zero the 8 slots
                const bool inr = (x >= g0) && (x < g11);
                float xr = (x - g0) * invh;             // >= 0 when inr
                int   m  = (int)xr;
                m = m < 0 ? 0 : (m > 10 ? 10 : m);
                float u   = xr - (float)m;
                float omu = 1.0f - u;
                float u2 = u * u, u3 = u2 * u;
                float w[4];
                w[0] = omu * omu * omu * (1.0f / 6.0f);            // j = m-3
                w[1] = (3.0f*u3 - 6.0f*u2 + 4.0f) * (1.0f/6.0f);   // j = m-2
                w[2] = (-3.0f*u3 + 3.0f*u2 + 3.0f*u + 1.0f) * (1.0f/6.0f); // m-1
                w[3] = u3 * (1.0f / 6.0f);                          // j = m
#pragma unroll
                for (int i = 0; i < 4; ++i) {
                    int  j  = m - 3 + i;
                    bool ok = inr && (j >= 0) && (j < 8);
                    bf16* p = ok ? (slot + j) : dummy;  // cndmask on address
                    *p = (bf16)w[i];
                }
            }
        }
        // ---------------- B tile: 256 cols x 32 k ---------------------------
        {
            const bf16* wp = Wc + (size_t)(n0 + tid) * KTOT + kb;
            v8bf w0 = *(const v8bf*)(wp);
            v8bf w1 = *(const v8bf*)(wp + 8);
            v8bf w2 = *(const v8bf*)(wp + 16);
            v8bf w3 = *(const v8bf*)(wp + 24);
            bf16* bq = &Bs[tid * LPAD];
            *(v8bf*)(bq)      = w0;
            *(v8bf*)(bq + 8)  = w1;
            *(v8bf*)(bq + 16) = w2;
            *(v8bf*)(bq + 24) = w3;
        }
        __syncthreads();

        // ---------------- WMMA: 2 (M) x 8 (N) tiles per wave ----------------
        v16bf afr[2];
#pragma unroll
        for (int a = 0; a < 2; ++a) {
            // ISA 16-bit A: lanes<16 hold K 0-7 & 16-23; lanes>=16 K 8-15 & 24-31
            const bf16* ap = &As[(wm * 32 + a * 16 + lr) * LPAD + half * 8];
            v8bf lo = *(const v8bf*)ap;
            v8bf hi = *(const v8bf*)(ap + 16);
#pragma unroll
            for (int q = 0; q < 8; ++q) { afr[a][q] = lo[q]; afr[a][q + 8] = hi[q]; }
        }
#pragma unroll
        for (int t = 0; t < 8; ++t) {
            // ISA 16-bit B: N = lane&15; lanes<16 hold K 0-7 & 16-23; lanes>=16 K 8-15 & 24-31
            const bf16* bp = &Bs[(wn * 128 + t * 16 + lr) * LPAD + half * 8];
            v8bf lo = *(const v8bf*)bp;
            v8bf hi = *(const v8bf*)(bp + 16);
            v16bf bfr;
#pragma unroll
            for (int q = 0; q < 8; ++q) { bfr[q] = lo[q]; bfr[q + 8] = hi[q]; }
#pragma unroll
            for (int a = 0; a < 2; ++a)
                acc[a][t] = __builtin_amdgcn_wmma_f32_16x16x32_bf16(
                    false, afr[a], false, bfr, (short)0, acc[a][t], false, false);
        }
        __syncthreads();
    }

    // ------- epilogue: transpose via LDS, coalesced stores along L ----------
#pragma unroll
    for (int p = 0; p < 4; ++p) {               // 4 passes of 64 output chans
        if (wn == (p >> 1)) {
            const int tb = (p & 1) * 4;
#pragma unroll
            for (int tt = 0; tt < 4; ++tt)
#pragma unroll
                for (int a = 0; a < 2; ++a)
#pragma unroll
                    for (int r = 0; r < 8; ++r) {
                        int nl = tt * 16 + lr;              // 0..63 in pass
                        int ml = wm * 32 + a * 16 + r + half * 8;
                        Cs[nl * CPAD + ml] = acc[a][tb + tt][r];
                    }
        }
        __syncthreads();
        const int o_rel = tid >> 2;           // 0..63
        const int lseg  = (tid & 3) * 32;
        float* op = Out + (((size_t)(b * OUT_F + n0 + p * 64 + o_rel)) << 11)
                        + l0 + lseg;
        const float* cp = &Cs[o_rel * CPAD + lseg];
#pragma unroll
        for (int v = 0; v < 8; ++v)
            *(v4f*)(op + v * 4) = *(const v4f*)(cp + v * 4);
        __syncthreads();
    }
}

extern "C" void kernel_launch(void* const* d_in, const int* in_sizes, int n_in,
                              void* d_out, int out_size, void* d_ws, size_t ws_size,
                              hipStream_t stream) {
    const float* x  = (const float*)d_in[0];   // (8,512,2048)
    const float* bw = (const float*)d_in[1];   // (512,512)
    const float* sw = (const float*)d_in[2];   // (512,512,8)
    const float* ss = (const float*)d_in[3];   // (512,512)
    const float* gr = (const float*)d_in[4];   // (512,12)
    float* out = (float*)d_out;                // (8,512,2048)
    bf16* Wc = (bf16*)d_ws;                    // 512*4608*2 = 4.7 MB

    int total = OUT_F * KTOT;
    kan_pack_weights<<<(total + 255) / 256, 256, 0, stream>>>(bw, sw, ss, Wc);

    dim3 grid(128, 2);                         // M tiles x N tiles (256 wide)
    kan_main<<<grid, 256, 0, stream>>>(x, gr, Wc, out);
}
